// GVAE_9878424781051
// MI455X (gfx1250) — matrix operand order repaired and checked by reading, not compiled, not run on hardware
//
#include <hip/hip_runtime.h>
#include <stdint.h>

// ---------------------------------------------------------------------------
// GVAE pipeline for MI455X (gfx1250, wave32, WMMA, async-LDS, pk-f16 atomics)
// d_out = [edge_score (E) | mu (N*64) | logstd (N*64)], all f32.
// ---------------------------------------------------------------------------

typedef __attribute__((ext_vector_type(16))) _Float16 v16h;
typedef __attribute__((ext_vector_type(8)))  float    v8f;

union Frag16 { v16h v; uint4 q[2]; };   // 32 bytes: one WMMA 16-bit A/B fragment

static __device__ __forceinline__ v8f wmma_f16(v16h a, v16h b, v8f c) {
  // (neg_a, A, neg_b, B, c_mod, C, reuse_a, reuse_b)
  return __builtin_amdgcn_wmma_f32_16x16x32_f16(false, a, false, b, (short)0, c,
                                                false, false);
}

// Issue one 16B async global->LDS copy (ASYNCcnt-tracked DMA, no VGPR round
// trip). ldsOff = wave-relative LDS byte offset, g = 64-bit global address.
static __device__ __forceinline__ void async_copy_b128(uint32_t ldsOff,
                                                       uint64_t g) {
  asm volatile("global_load_async_to_lds_b128 %0, %1, off"
               :: "v"(ldsOff), "v"(g) : "memory");
}
static __device__ __forceinline__ void async_wait0() {
  asm volatile("s_wait_asynccnt 0x0" ::: "memory");
}

// Packed f16 atomic add (GLOBAL_ATOMIC_PK_ADD_F16, no return -> STOREcnt).
static __device__ __forceinline__ void pk_atomic_add_f16(uint64_t addr,
                                                         uint32_t pk) {
  asm volatile("global_atomic_pk_add_f16 %0, %1, off"
               :: "v"(addr), "v"(pk) : "memory");
}

static __device__ __forceinline__ uint32_t lds_off(const void* p) {
  // Generic LDS pointer: high 32 bits = SHARED aperture, low 32 = LDS offset.
  return (uint32_t)(uintptr_t)p;
}

// Counter-based gaussian (Box-Muller). NOTE: JAX threefry key(42) is not
// reproducible here; deterministic per-(node,col) hash stands in for eps.
static __device__ __forceinline__ float hash_gauss(uint32_t a, uint32_t b) {
  uint32_t s = a * 0x9E3779B9u ^ b * 0x85EBCA6Bu;
  s ^= s >> 16; s *= 0x7FEB352Du; s ^= s >> 15; s *= 0x846CA68Bu; s ^= s >> 16;
  uint32_t t = s * 0x27D4EB2Fu + 0x165667B1u;
  t ^= t >> 15; t *= 0x2C1B3C6Du; t ^= t >> 12;
  float u1 = (float)((s >> 8) + 1u) * (1.0f / 16777217.0f);
  float u2 = (float)(t >> 8) * (1.0f / 16777216.0f);
  float r = sqrtf(-2.0f * __logf(u1));
  return r * __cosf(6.2831853071795864f * u2);
}

// --- K0: deg = 1 (self loop) ------------------------------------------------
__global__ void gvae_k0_deg_init(float* __restrict__ deg, int N) {
  int i = blockIdx.x * blockDim.x + threadIdx.x;
  if (i < N) deg[i] = 1.0f;
}

// --- K1: deg[dst[e]] += 1 ---------------------------------------------------
__global__ void gvae_k1_deg_count(const int* __restrict__ dst,
                                  float* __restrict__ deg, int E) {
  int e = blockIdx.x * blockDim.x + threadIdx.x;
  if (e >= E) return;
  if ((e & 63) == 0) __builtin_prefetch(dst + e + 4096, 0, 0);
  atomicAdd(&deg[dst[e]], 1.0f);
}

// --- Kw: one-time weight pre-convert f32 -> f16 (WMMA B layout = row major) -
__global__ void gvae_kw_weights(const float* __restrict__ mu_w,
                                const float* __restrict__ ls_w,
                                const float* __restrict__ dec_w1,
                                _Float16* __restrict__ wmuH,
                                _Float16* __restrict__ wlsH,
                                _Float16* __restrict__ w1H) {
  int i = blockIdx.x * blockDim.x + threadIdx.x;
  if (i < 4096) wmuH[i] = (_Float16)mu_w[i];
  else if (i < 8192) wlsH[i - 4096] = (_Float16)ls_w[i - 4096];
  else if (i < 16384) w1H[i - 8192] = (_Float16)dec_w1[i - 8192];
}

// --- K2: h = x@enc_w (f16) ; dinv = rsqrt(deg) ; agg = enc_b + h*dinv^2 -----
__global__ __launch_bounds__(256)
void gvae_k2_transform(const float* __restrict__ x,
                       const float* __restrict__ enc_w,
                       const float* __restrict__ enc_b,
                       const float* __restrict__ deg,
                       float* __restrict__ dinv,
                       _Float16* __restrict__ hh,
                       _Float16* __restrict__ aggh, int N) {
  int gid = blockIdx.x * 256 + threadIdx.x;
  int node = gid >> 6;
  if (node >= N) return;
  int k = gid & 63;
  float x0 = x[(size_t)node * 2 + 0];
  float x1 = x[(size_t)node * 2 + 1];
  float di = rsqrtf(deg[node]);
  if (k == 0) dinv[node] = di;
  float hv = x0 * enc_w[k] + x1 * enc_w[64 + k];
  hh[(size_t)node * 64 + k] = (_Float16)hv;                    // f16 gather src
  aggh[(size_t)node * 64 + k] = (_Float16)(enc_b[k] + hv * di * di);
}

// --- K3: scatter  agg[dst] += h[src] * dinv[src]*dinv[dst] ------------------
// 8 lanes/edge, 8 halves per lane; packed-f16 atomics halve atomic op count
// (32 ops/edge vs 64) and land in the 192MB L2 where agg (12.8MB) resides.
__global__ __launch_bounds__(256)
void gvae_k3_scatter(const int* __restrict__ src, const int* __restrict__ dst,
                     const float* __restrict__ dinv,
                     const _Float16* __restrict__ hh,
                     _Float16* __restrict__ aggh, int E) {
  int gid = blockIdx.x * 256 + threadIdx.x;
  int e = gid >> 3;
  if (e >= E) return;
  if ((gid & 7) == 0) {
    __builtin_prefetch(src + e + 4096, 0, 0);
    __builtin_prefetch(dst + e + 4096, 0, 0);
  }
  int q = (gid & 7) * 8;                           // half index within row
  int s = src[e], d = dst[e];
  float norm = dinv[s] * dinv[d];
  union { uint4 v; _Float16 h[8]; } u;
  u.v = *(const uint4*)(hh + (size_t)s * 64 + q);  // 16B f16 gather
  uint64_t base = (uint64_t)(uintptr_t)(aggh + (size_t)d * 64 + q);
#pragma unroll
  for (int j = 0; j < 4; ++j) {
    union { uint32_t u32; _Float16 h[2]; } pk;
    pk.h[0] = (_Float16)((float)u.h[2 * j + 0] * norm);
    pk.h[1] = (_Float16)((float)u.h[2 * j + 1] * norm);
    pk_atomic_add_f16(base + (uint64_t)(j * 4), pk.u32);
  }
}

// --- K4: mu/logstd GEMMs (WMMA f16) + reparameterize ------------------------
// Block = 256 thr = 8 waves; each wave owns a 16-node tile (128 nodes/block).
// All f16 operands staged via async global->LDS DMA (raw byte copy).
__global__ __launch_bounds__(256)
void gvae_k4_mu_ls(const _Float16* __restrict__ aggh,
                   const _Float16* __restrict__ wmuH,
                   const _Float16* __restrict__ wlsH,
                   const float* __restrict__ mu_b, const float* __restrict__ ls_b,
                   float* __restrict__ mu_out, float* __restrict__ ls_out,
                   _Float16* __restrict__ zh, int N) {
  __shared__ _Float16 aS[128 * 64];     // 16 KB: A tile (f16)
  __shared__ _Float16 wmuS[64 * 64];    //  8 KB
  __shared__ _Float16 wlsS[64 * 64];    //  8 KB
  __shared__ float bmuS[64], blsS[64];

  const int tid = threadIdx.x;
  const int node0 = blockIdx.x * 128;

  // Async DMA staging: aggh tile (tail blocks read into f16 ws slack; stores
  // below are guarded), plus the two pre-converted f16 weight blobs.
  {
    const uint32_t aL  = lds_off(aS);
    const uint32_t muL = lds_off(wmuS);
    const uint32_t lsL = lds_off(wlsS);
    const _Float16* aG = aggh + (size_t)node0 * 64;
    for (int i = tid; i < 1024; i += 256)          // 16 KB A tile
      async_copy_b128(aL + (uint32_t)(i * 16),
                      (uint64_t)(uintptr_t)(aG + (size_t)i * 8));
    for (int i = tid; i < 512; i += 256) {         // 8 KB + 8 KB weights
      async_copy_b128(muL + (uint32_t)(i * 16),
                      (uint64_t)(uintptr_t)(wmuH + (size_t)i * 8));
      async_copy_b128(lsL + (uint32_t)(i * 16),
                      (uint64_t)(uintptr_t)(wlsH + (size_t)i * 8));
    }
    if (tid < 64) { bmuS[tid] = mu_b[tid]; blsS[tid] = ls_b[tid]; }
    async_wait0();
  }
  __syncthreads();

  const int wid = tid >> 5, lane = tid & 31;
  const int mrow0 = wid * 16;
  const int m = lane & 15;
  const int sel = (lane < 16) ? 0 : 1;          // ISA A-layout lane-half select

  // A fragments: K=0..31 and K=32..63 (16-bit A layout, §7.12.2)
  Frag16 a0, a1;
  const uint4* ar = (const uint4*)(aS + (size_t)(mrow0 + m) * 64);
  a0.q[0] = ar[0 + sel]; a0.q[1] = ar[2 + sel];
  a1.q[0] = ar[4 + sel]; a1.q[1] = ar[6 + sel];

  v8f cmu[4], cls[4];
#pragma unroll
  for (int nt = 0; nt < 4; ++nt) {
    Frag16 b0, b1;
    // B layout: lane holds N=lane%16; elements hold K in A's (table) order
    const _Float16* w0 = wmuS + nt * 16 + m;
    const _Float16* w1 = w0 + 32 * 64;
#pragma unroll
    for (int i = 0; i < 16; ++i) {
      int kk = (i & 7) + 8 * sel + 16 * (i >> 3);
      b0.v[i] = w0[kk * 64]; b1.v[i] = w1[kk * 64];
    }
    v8f c = {};
    c = wmma_f16(a0.v, b0.v, c);
    c = wmma_f16(a1.v, b1.v, c);
    cmu[nt] = c;

    const _Float16* v0 = wlsS + nt * 16 + m;
    const _Float16* v1 = v0 + 32 * 64;
#pragma unroll
    for (int i = 0; i < 16; ++i) {
      int kk = (i & 7) + 8 * sel + 16 * (i >> 3);
      b0.v[i] = v0[kk * 64]; b1.v[i] = v1[kk * 64];
    }
    v8f d = {};
    d = wmma_f16(a0.v, b0.v, d);
    d = wmma_f16(a1.v, b1.v, d);
    cls[nt] = d;
  }

  // C/D layout: VGPR r, lanes 0-15 -> M=r, lanes 16-31 -> M=8+r; N=lane%16
  const int half = lane >> 4;
#pragma unroll
  for (int nt = 0; nt < 4; ++nt) {
    int col = nt * 16 + (lane & 15);
    float bm = bmuS[col], bl = blsS[col];
#pragma unroll
    for (int r = 0; r < 8; ++r) {
      int node = node0 + mrow0 + r + half * 8;
      float muv = cmu[nt][r] + bm;
      float lsv = cls[nt][r] + bl;
      float eps = hash_gauss((uint32_t)node, (uint32_t)col);
      float z = muv + eps * __expf(0.5f * lsv);
      if (node < N) {
        mu_out[(size_t)node * 64 + col] = muv;
        ls_out[(size_t)node * 64 + col] = lsv;
        zh[(size_t)node * 64 + col] = (_Float16)z;   // f16 z halves gather BW
      }
    }
  }
}

// --- K5: decode: gather z[src]||z[dst], GEMM w/ dec_w1, relu, dot dec_w2 ----
// Block = 8 waves; each wave owns 16 edges (128 edges/block).
__global__ __launch_bounds__(256)
void gvae_k5_decode(const _Float16* __restrict__ zh,
                    const int* __restrict__ src, const int* __restrict__ dst,
                    const _Float16* __restrict__ w1H,
                    const float* __restrict__ dec_b1,
                    const float* __restrict__ dec_w2,
                    const float* __restrict__ dec_b2,
                    float* __restrict__ score, int E) {
  __shared__ _Float16 wS[128 * 64];     // 16 KB f16 dec_w1
  __shared__ float b1S[64], w2S[64];

  const int tid = threadIdx.x;
  {
    const uint32_t wL = lds_off(wS);
    for (int i = tid; i < 1024; i += 256)          // 16 KB async DMA
      async_copy_b128(wL + (uint32_t)(i * 16),
                      (uint64_t)(uintptr_t)(w1H + (size_t)i * 8));
    if (tid < 64) { b1S[tid] = dec_b1[tid]; w2S[tid] = dec_w2[tid]; }
    async_wait0();
  }
  __syncthreads();

  const int wid = tid >> 5, lane = tid & 31;
  const int edge0 = blockIdx.x * 128 + wid * 16;
  int e = edge0 + (lane & 15);
  if (e > E - 1) e = E - 1;                      // clamp (stores guarded)
  if ((lane & 15) == 0) {
    __builtin_prefetch(src + e + 2048, 0, 0);
    __builtin_prefetch(dst + e + 2048, 0, 0);
  }
  const int s = src[e], d = dst[e];
  const int sel = (lane < 16) ? 0 : 1;

  // A = z_cat row (K=0..127): frags from z[src] (K 0..63), z[dst] (K 64..127)
  Frag16 a0, a1, a2, a3;
  const uint4* ps = (const uint4*)(zh + (size_t)s * 64);
  const uint4* pd = (const uint4*)(zh + (size_t)d * 64);
  a0.q[0] = ps[0 + sel]; a0.q[1] = ps[2 + sel];
  a1.q[0] = ps[4 + sel]; a1.q[1] = ps[6 + sel];
  a2.q[0] = pd[0 + sel]; a2.q[1] = pd[2 + sel];
  a3.q[0] = pd[4 + sel]; a3.q[1] = pd[6 + sel];

  float p[8] = {0.f, 0.f, 0.f, 0.f, 0.f, 0.f, 0.f, 0.f};
#pragma unroll
  for (int nt = 0; nt < 4; ++nt) {
    v8f c = {};
    Frag16 b;
    const _Float16* wc = wS + nt * 16 + (lane & 15);
#pragma unroll
    for (int i = 0; i < 16; ++i) b.v[i] = wc[(0 * 32 + (i & 7) + 8 * sel + 16 * (i >> 3)) * 64];
    c = wmma_f16(a0.v, b.v, c);
#pragma unroll
    for (int i = 0; i < 16; ++i) b.v[i] = wc[(1 * 32 + (i & 7) + 8 * sel + 16 * (i >> 3)) * 64];
    c = wmma_f16(a1.v, b.v, c);
#pragma unroll
    for (int i = 0; i < 16; ++i) b.v[i] = wc[(2 * 32 + (i & 7) + 8 * sel + 16 * (i >> 3)) * 64];
    c = wmma_f16(a2.v, b.v, c);
#pragma unroll
    for (int i = 0; i < 16; ++i) b.v[i] = wc[(3 * 32 + (i & 7) + 8 * sel + 16 * (i >> 3)) * 64];
    c = wmma_f16(a3.v, b.v, c);

    int col = nt * 16 + (lane & 15);
    float bb = b1S[col], ww = w2S[col];
#pragma unroll
    for (int r = 0; r < 8; ++r) {
      float hv = c[r] + bb;
      hv = hv > 0.f ? hv : 0.f;                  // relu
      p[r] += hv * ww;                           // dot with dec_w2 column part
    }
  }

  // reduce over the 16 lanes of each half (N dimension)
#pragma unroll
  for (int r = 0; r < 8; ++r) {
#pragma unroll
    for (int off = 1; off < 16; off <<= 1)
      p[r] += __shfl_xor(p[r], off, 32);
  }

  if ((lane & 15) == 0) {
    const float b2 = dec_b2[0];
    const int half = lane >> 4;
#pragma unroll
    for (int r = 0; r < 8; ++r) {
      int ee = edge0 + r + half * 8;
      if (ee < E) score[ee] = 1.0f / (1.0f + __expf(-(p[r] + b2)));
    }
  }
}

// ---------------------------------------------------------------------------
extern "C" void kernel_launch(void* const* d_in, const int* in_sizes, int n_in,
                              void* d_out, int out_size, void* d_ws,
                              size_t ws_size, hipStream_t stream) {
  const float* x      = (const float*)d_in[0];
  const int*   ei     = (const int*)  d_in[1];
  const float* enc_w  = (const float*)d_in[2];
  const float* enc_b  = (const float*)d_in[3];
  const float* mu_w   = (const float*)d_in[4];
  const float* mu_b   = (const float*)d_in[5];
  const float* ls_w   = (const float*)d_in[6];
  const float* ls_b   = (const float*)d_in[7];
  const float* dec_w1 = (const float*)d_in[8];
  const float* dec_b1 = (const float*)d_in[9];
  const float* dec_w2 = (const float*)d_in[10];
  const float* dec_b2 = (const float*)d_in[11];

  const int N = in_sizes[0] / 2;     // x is [N, 2]
  const int E = in_sizes[1] / 2;     // edge_index is [2, E]
  const int* srcI = ei;
  const int* dstI = ei + E;

  // Workspace layout, 16B-aligned blocks.
  size_t Np = ((size_t)N + 3) & ~(size_t)3;
  float* ws   = (float*)d_ws;
  float* deg  = ws;
  float* dinv = ws + Np;
  _Float16* wmuH = (_Float16*)(ws + 2 * Np);      // 4096 halves
  _Float16* wlsH = wmuH + 4096;                   // 4096 halves
  _Float16* w1H  = wlsH + 4096;                   // 8192 halves
  _Float16* hh   = w1H + 8192;                    // N*64 halves
  _Float16* aggh = hh + (size_t)N * 64;           // N*64 halves
  _Float16* zh   = aggh + (size_t)N * 64;         // N*64 halves (also gives
                                                  // aggh tail-read slack)
  float* scoreO = (float*)d_out;
  float* muO    = scoreO + (size_t)E;
  float* lsO    = muO + (size_t)N * 64;

  gvae_k0_deg_init<<<(N + 255) / 256, 256, 0, stream>>>(deg, N);
  gvae_k1_deg_count<<<(E + 255) / 256, 256, 0, stream>>>(dstI, deg, E);
  gvae_kw_weights<<<64, 256, 0, stream>>>(mu_w, ls_w, dec_w1, wmuH, wlsH, w1H);
  gvae_k2_transform<<<((size_t)N * 64 + 255) / 256, 256, 0, stream>>>(
      x, enc_w, enc_b, deg, dinv, hh, aggh, N);
  gvae_k3_scatter<<<((size_t)E * 8 + 255) / 256, 256, 0, stream>>>(
      srcI, dstI, dinv, hh, aggh, E);
  gvae_k4_mu_ls<<<(N + 127) / 128, 256, 0, stream>>>(
      aggh, wmuH, wlsH, mu_b, ls_b, muO, lsO, zh, N);
  gvae_k5_decode<<<(E + 127) / 128, 256, 0, stream>>>(
      zh, srcI, dstI, w1H, dec_b1, dec_w2, dec_b2, scoreO, E);
}
